// CharacterLoss_3977139716295
// MI455X (gfx1250) — matrix operand order repaired and checked by reading, not compiled, not run on hardware
//
#include <hip/hip_runtime.h>

// ---------------------------------------------------------------------------
// CDNA5 / gfx1250 fused cosine-sim + BCEWithLogits loss.
//   dots = X X^T via v_wmma_f32_16x16x32_bf16 with split-bf16 (hi/lo)
//   compensation: dot ~= hi*hi + hi*lo + lo*hi  (≈ fp32 accuracy)
//   Double-buffered LDS tiles filled by global_load_async_to_lds_b128
//   (ASYNCcnt path) so DMA overlaps WMMA compute.
// ---------------------------------------------------------------------------

typedef __attribute__((ext_vector_type(16))) __bf16 v16bf;
typedef __attribute__((ext_vector_type(8)))  float  v8f;

#define D_DIM   4096
#define DU      (D_DIM / 2)   // uints (2 x bf16) per row
#define DU4     (DU / 4)      // 16-byte chunks per row
#define K_STEP  32
#define N_ITERS (D_DIM / K_STEP)
#define COS_EPS 1e-8f

struct __attribute__((aligned(16))) U4 { unsigned int x, y, z, w; };
union Frag { U4 q[2]; v16bf v; };

__device__ __forceinline__ unsigned short f32_to_bf16_rne(float f) {
    unsigned u = __float_as_uint(f);
    u += 0x7FFFu + ((u >> 16) & 1u);
    return (unsigned short)(u >> 16);
}
__device__ __forceinline__ float bf16_to_f32(unsigned short h) {
    return __uint_as_float(((unsigned)h) << 16);
}

// LDS byte offset of a __shared__ object (addrspace(3) is a 32-bit offset).
__device__ __forceinline__ unsigned lds_off_u(const void* p) {
    return (unsigned)(unsigned long long)
           (__attribute__((address_space(3))) const void*)p;
}

// Async DMA: 16 bytes per lane, global -> LDS, tracked by ASYNCcnt.
__device__ __forceinline__ void async_ld16(unsigned lds_byte_off,
                                           const void* gsrc) {
    asm volatile("global_load_async_to_lds_b128 %0, %1, off"
                 :: "v"(lds_byte_off), "v"((unsigned long long)gsrc)
                 : "memory");
}
__device__ __forceinline__ void wait_async0() {
    asm volatile("s_wait_asynccnt 0x0" ::: "memory");
}

// ---------------------------------------------------------------------------
// Kernel 1: gather rows by token_indices, split into bf16 hi/lo planes,
// compute per-row L2 norms. One block per gathered row.
// ---------------------------------------------------------------------------
__global__ void __launch_bounds__(256)
gather_split_kernel(const float* __restrict__ data,
                    const int*   __restrict__ tok,
                    unsigned int* __restrict__ Xhi,
                    unsigned int* __restrict__ Xlo,
                    float* __restrict__ norms)
{
    __shared__ float red[256];
    const int row = blockIdx.x;
    const int tid = threadIdx.x;

    const float2* src = (const float2*)(data + (size_t)tok[row] * D_DIM);
    unsigned int* hi  = Xhi + (size_t)row * DU;
    unsigned int* lo  = Xlo + (size_t)row * DU;

    float acc = 0.f;
#pragma unroll 4
    for (int p = tid; p < DU; p += 256) {
        float2 v = src[p];
        unsigned short h0 = f32_to_bf16_rne(v.x);
        unsigned short h1 = f32_to_bf16_rne(v.y);
        unsigned short l0 = f32_to_bf16_rne(v.x - bf16_to_f32(h0));
        unsigned short l1 = f32_to_bf16_rne(v.y - bf16_to_f32(h1));
        hi[p] = (unsigned)h0 | ((unsigned)h1 << 16);
        lo[p] = (unsigned)l0 | ((unsigned)l1 << 16);
        acc += v.x * v.x + v.y * v.y;
    }
    red[tid] = acc;
    __syncthreads();
    for (int s = 128; s > 0; s >>= 1) {
        if (tid < s) red[tid] += red[tid + s];
        __syncthreads();
    }
    if (tid == 0) norms[row] = sqrtf(red[0]);
}

// ---------------------------------------------------------------------------
// Kernel 2: 128x128 output tile per block (upper block-triangle only),
// 8 wave32s in a 2(M) x 4(N) grid; each wave owns 4x2 16x16 WMMA subtiles.
// Double-buffered async global->LDS staging overlapped with WMMA compute.
// Fused epilogue: cosine normalize + BCEWithLogits + block tree-reduction.
// ---------------------------------------------------------------------------
__global__ void __launch_bounds__(256)
cosbce_gemm_kernel(const unsigned int* __restrict__ Xhi,
                   const unsigned int* __restrict__ Xlo,
                   const float* __restrict__ norms,
                   const int*   __restrict__ names,
                   float* __restrict__ partials,
                   int nbt, float invK2)
{
    __shared__ __align__(16) unsigned int sAhi[2][128 * 16];
    __shared__ __align__(16) unsigned int sAlo[2][128 * 16];
    __shared__ __align__(16) unsigned int sBhi[2][128 * 16];
    __shared__ __align__(16) unsigned int sBlo[2][128 * 16];
    __shared__ float sNormM[128], sNormN[128];
    __shared__ int   sNameM[128], sNameN[128];
    __shared__ float sRed[256];

    // Decode upper-triangle block coordinates from linear block index.
    int bm = 0, rem = blockIdx.x;
    while (rem >= nbt - bm) { rem -= nbt - bm; bm++; }
    const int bn = bm + rem;
    const float weight = (bm == bn) ? 1.f : 2.f;

    const int tid  = threadIdx.x;
    const int lane = tid & 31;
    const int wave = tid >> 5;
    const int wm   = wave >> 2;   // 0..1 : 64 rows of M each
    const int wn   = wave & 3;    // 0..3 : 32 cols of N each
    const int l15  = lane & 15;
    const int g    = lane >> 4;

    // Stage norms / name ids for this block's row & col ranges.
    if (tid < 128) {
        sNormM[tid] = norms[bm * 128 + tid];
        sNameM[tid] = names[bm * 128 + tid];
    } else {
        int i = tid - 128;
        sNormN[i] = norms[bn * 128 + i];
        sNameN[i] = names[bn * 128 + i];
    }

    // LDS byte offsets for async DMA destinations.
    const unsigned offAhi[2] = { lds_off_u(&sAhi[0][0]), lds_off_u(&sAhi[1][0]) };
    const unsigned offAlo[2] = { lds_off_u(&sAlo[0][0]), lds_off_u(&sAlo[1][0]) };
    const unsigned offBhi[2] = { lds_off_u(&sBhi[0][0]), lds_off_u(&sBhi[1][0]) };
    const unsigned offBlo[2] = { lds_off_u(&sBlo[0][0]), lds_off_u(&sBlo[1][0]) };

    const char* gXhi = (const char*)Xhi;
    const char* gXlo = (const char*)Xlo;
    const size_t rowM0 = (size_t)(bm * 128);
    const size_t rowN0 = (size_t)(bn * 128);

    // Issue one 128x32 (x4 planes) tile prefetch into buffer `buf`.
    auto stage_async = [&](int buf, int it) {
        const int k4 = it * (K_STEP / 8);   // 16B chunk offset within a row
#pragma unroll
        for (int q = tid; q < 512; q += 256) {
            const int r = q >> 2, c = q & 3;
            const size_t gm = ((rowM0 + r) * DU4 + k4 + c) * 16;
            const size_t gn = ((rowN0 + r) * DU4 + k4 + c) * 16;
            const unsigned loff = (unsigned)q * 16u;
            async_ld16(offAhi[buf] + loff, gXhi + gm);
            async_ld16(offAlo[buf] + loff, gXlo + gm);
            async_ld16(offBhi[buf] + loff, gXhi + gn);
            async_ld16(offBlo[buf] + loff, gXlo + gn);
        }
    };

    v8f acc[4][2];
    const v8f vzero = {0.f, 0.f, 0.f, 0.f, 0.f, 0.f, 0.f, 0.f};
#pragma unroll
    for (int mi = 0; mi < 4; mi++)
#pragma unroll
        for (int ni = 0; ni < 2; ni++) acc[mi][ni] = vzero;

    stage_async(0, 0);   // prologue prefetch

    for (int it = 0; it < N_ITERS; ++it) {
        const int cur = it & 1;
        wait_async0();       // my DMA writes into buf `cur` have landed
        __syncthreads();     // everyone's landed; everyone done reading 1-cur
        if (it + 1 < N_ITERS) stage_async(1 - cur, it + 1);

        const U4* pAhi = (const U4*)sAhi[cur];
        const U4* pAlo = (const U4*)sAlo[cur];
        const U4* pBhi = (const U4*)sBhi[cur];
        const U4* pBlo = (const U4*)sBlo[cur];

#pragma unroll
        for (int ni = 0; ni < 2; ni++) {
            // B fragment (32x16): lane holds N=l15, K = g*16 .. g*16+15.
            const int nr = (wn * 32 + ni * 16 + l15) * 4;
            Frag bh, bl;
            bh.q[0] = pBhi[nr + g * 2];  bh.q[1] = pBhi[nr + g * 2 + 1];
            bl.q[0] = pBlo[nr + g * 2];  bl.q[1] = pBlo[nr + g * 2 + 1];
#pragma unroll
            for (int mi = 0; mi < 4; mi++) {
                // A fragment (16x32): lane holds M=l15;
                // K = g*16 .. g*16+15.
                const int mr = (wm * 64 + mi * 16 + l15) * 4;
                Frag ah, al;
                ah.q[0] = pAhi[mr + g * 2];  ah.q[1] = pAhi[mr + g * 2 + 1];
                al.q[0] = pAlo[mr + g * 2];  al.q[1] = pAlo[mr + g * 2 + 1];

                v8f c = acc[mi][ni];
                c = __builtin_amdgcn_wmma_f32_16x16x32_bf16(
                        false, ah.v, false, bh.v, (short)0, c, false, false);
                c = __builtin_amdgcn_wmma_f32_16x16x32_bf16(
                        false, ah.v, false, bl.v, (short)0, c, false, false);
                c = __builtin_amdgcn_wmma_f32_16x16x32_bf16(
                        false, al.v, false, bh.v, (short)0, c, false, false);
                acc[mi][ni] = c;
            }
        }
        __syncthreads();     // done reading buf `cur` before it is refilled
    }

    // Fused epilogue: cosine normalize, BCEWithLogits, local accumulate.
    // C/D layout: element i of v8f -> M = i + (lane>>4)*8, N = lane&15.
    float lsum = 0.f;
    const int mhi = g * 8;
#pragma unroll
    for (int mi = 0; mi < 4; mi++) {
#pragma unroll
        for (int ni = 0; ni < 2; ni++) {
            const int nl = wn * 32 + ni * 16 + l15;
            const float nn = sNormN[nl];
            const int   idn = sNameN[nl];
#pragma unroll
            for (int i = 0; i < 8; i++) {
                const int ml = wm * 64 + mi * 16 + mhi + i;
                const float denom = fmaxf(sNormM[ml] * nn, COS_EPS);
                const float x = acc[mi][ni][i] / denom;
                const float tgt = (sNameM[ml] == idn) ? 1.f : 0.f;
                lsum += fmaxf(x, 0.f) - x * tgt + log1pf(expf(-fabsf(x)));
            }
        }
    }

    sRed[tid] = lsum;
    __syncthreads();
    for (int s = 128; s > 0; s >>= 1) {
        if (tid < s) sRed[tid] += sRed[tid + s];
        __syncthreads();
    }
    if (tid == 0) partials[blockIdx.x] = sRed[0] * invK2 * weight;
}

// ---------------------------------------------------------------------------
// Kernel 3: deterministic fixed-order sum of block partials -> scalar loss.
// ---------------------------------------------------------------------------
__global__ void __launch_bounds__(256)
final_reduce_kernel(const float* __restrict__ partials, int n,
                    float* __restrict__ out)
{
    __shared__ float red[256];
    const int tid = threadIdx.x;
    float s = 0.f;
    for (int i = tid; i < n; i += 256) s += partials[i];
    red[tid] = s;
    __syncthreads();
    for (int k = 128; k > 0; k >>= 1) {
        if (tid < k) red[tid] += red[tid + k];
        __syncthreads();
    }
    if (tid == 0) out[0] = red[0];
}

// ---------------------------------------------------------------------------
extern "C" void kernel_launch(void* const* d_in, const int* in_sizes, int n_in,
                              void* d_out, int out_size, void* d_ws, size_t ws_size,
                              hipStream_t stream)
{
    const float* data  = (const float*)d_in[0];
    const int*   tok   = (const int*)d_in[1];
    const int*   names = (const int*)d_in[2];
    const int K = in_sizes[1];            // 2048

    // Workspace layout: Xhi | Xlo (bf16 planes as uint) | norms | partials.
    unsigned int* Xhi   = (unsigned int*)d_ws;
    unsigned int* Xlo   = Xhi + (size_t)K * DU;
    float* norms        = (float*)(Xlo + (size_t)K * DU);
    float* partials     = norms + K;

    const int nbt     = K / 128;                 // 16 block tiles per dim
    const int nblocks = nbt * (nbt + 1) / 2;     // upper triangle: 136
    const float invK2 = 1.0f / ((float)K * (float)K);

    gather_split_kernel<<<K, 256, 0, stream>>>(data, tok, Xhi, Xlo, norms);
    cosbce_gemm_kernel<<<nblocks, 256, 0, stream>>>(Xhi, Xlo, norms, names,
                                                    partials, nbt, invK2);
    final_reduce_kernel<<<1, 256, 0, stream>>>(partials, nblocks, (float*)d_out);
}
